// DenseRNN_81423989997661
// MI455X (gfx1250) — compile-verified, matrix operand order checked by reading, not run on hardware
//
#include <hip/hip_runtime.h>
#include <hip/hip_bf16.h>
#include <math.h>

typedef __attribute__((ext_vector_type(16))) _Float16 v16h;
typedef __attribute__((ext_vector_type(8)))  _Float16 v8h;
typedef __attribute__((ext_vector_type(8)))  float    v8f;

#define RNN_B 256
#define RNN_T 1024
#define RNN_D 128
#define RNN_H 128
#define LDS_STRIDE 136   // 128 + 8 halves pad -> rows spread across bank groups

// Native CDNA5 tanh (V_TANH_F32) if the toolchain exposes it; else inline asm.
__device__ __forceinline__ float fast_tanh(float v) {
#if __has_builtin(__builtin_amdgcn_tanhf)
    return __builtin_amdgcn_tanhf(v);
#else
    float r;
    asm volatile("v_tanh_f32 %0, %1" : "=v"(r) : "v"(v));
    return r;
#endif
}

// One workgroup = 16 batch rows, 8 waves (wave32). Wave n owns output columns
// [16n, 16n+16). Weights stay in registers as f16 WMMA B-fragments; the 16x128
// state tile lives in LDS as f16 and is re-read as WMMA A-fragments each step.
__global__ __launch_bounds__(256)
void DenseRNN_elman_wmma_kernel(const float* __restrict__ x,      // [B,T,D]
                                const float* __restrict__ w,      // [D,H]
                                const float* __restrict__ sw,     // [H,H]
                                const float* __restrict__ bias,   // [H]
                                float* __restrict__ out)          // [B,H]
{
    __shared__ __align__(16) _Float16 sX[16 * LDS_STRIDE]; // x tile (f16) for current t
    __shared__ __align__(16) _Float16 sS[16 * LDS_STRIDE]; // state tile (f16)

    const int tid    = threadIdx.x;
    const int wave   = tid >> 5;          // 0..7  -> column tile
    const int lane   = tid & 31;
    const int m      = lane & 15;         // matrix row within tile (A) / col (B,C)
    const int kh     = lane >> 4;         // K-half selector per ISA A/B layout
    const int wgbase = blockIdx.x * 16;   // first batch row of this workgroup
    const int col    = wave * 16 + m;     // global H column owned by this lane

    // ---- Load persistent B-fragments of w and state_weight (f32 -> f16) ----
    // B layout (32x16 f16): lane holds column `col`; halves j=0..7 are
    // K = k0*32 + kh*8 + j, halves j=8..15 are K = k0*32 + 16 + kh*8 + (j-8).
    v16h Win[4], Wrec[4];
#pragma unroll
    for (int k0 = 0; k0 < 4; ++k0) {
        v16h bw, bs;
#pragma unroll
        for (int j = 0; j < 16; ++j) {
            const int kx = k0 * 32 + ((j >= 8) ? 16 : 0) + kh * 8 + (j & 7);
            bw[j] = (_Float16)w [kx * RNN_H + col];
            bs[j] = (_Float16)sw[kx * RNN_H + col];
        }
        Win[k0] = bw;
        Wrec[k0] = bs;
    }
    const float bcol = bias[col];   // one bias scalar per lane (its column)

    // x-tile staging assignment: 256 threads x 8 floats = 16x128 tile
    const int xrow = tid >> 4;            // 0..15
    const int xcol = (tid & 15) * 8;      // 0,8,...,120
    const float* xrow_ptr = x + ((size_t)(wgbase + xrow) * RNN_T) * RNN_D + xcol;

    v8f st = {};   // this wave's current state tile (f32, C/D layout)

    // Software pipeline: global loads for step t+1 issue before step t's WMMAs.
    float4 a0 = ((const float4*)xrow_ptr)[0];
    float4 a1 = ((const float4*)xrow_ptr)[1];

    for (int t = 0; t < RNN_T; ++t) {
        // ---- Stage x[:, t, :] tile into LDS as f16 (from pipelined regs) ----
        {
            v8h hx;
            hx[0] = (_Float16)a0.x; hx[1] = (_Float16)a0.y;
            hx[2] = (_Float16)a0.z; hx[3] = (_Float16)a0.w;
            hx[4] = (_Float16)a1.x; hx[5] = (_Float16)a1.y;
            hx[6] = (_Float16)a1.z; hx[7] = (_Float16)a1.w;
            *(v8h*)&sX[xrow * LDS_STRIDE + xcol] = hx;
        }
        // Issue next step's loads now; latency hides behind the WMMAs below.
        {
            const int tn = (t + 1 < RNN_T) ? (t + 1) : t;
            const float4* pn = (const float4*)(xrow_ptr + (size_t)tn * RNN_D);
            a0 = pn[0];
            a1 = pn[1];
        }
        __syncthreads();   // sX visible; previous step's sS writes visible

        // ---- c = bias; c += x_tile @ w  (4 x K=32 chunks), f32 accumulate ----
        v8f c;
#pragma unroll
        for (int r = 0; r < 8; ++r) c[r] = bcol;
#pragma unroll
        for (int k0 = 0; k0 < 4; ++k0) {
            const v8h lo = *(const v8h*)&sX[m * LDS_STRIDE + k0 * 32 + kh * 8];
            const v8h hi = *(const v8h*)&sX[m * LDS_STRIDE + k0 * 32 + 16 + kh * 8];
            v16h a;
#pragma unroll
            for (int i = 0; i < 8; ++i) { a[i] = lo[i]; a[i + 8] = hi[i]; }
            c = __builtin_amdgcn_wmma_f32_16x16x32_f16(
                    false, a, false, Win[k0], (short)0, c, false, false);
        }

        // ---- + state @ state_weight (skipped at t==0 per reference) ----
        if (t > 0) {
#pragma unroll
            for (int k0 = 0; k0 < 4; ++k0) {
                const v8h lo = *(const v8h*)&sS[m * LDS_STRIDE + k0 * 32 + kh * 8];
                const v8h hi = *(const v8h*)&sS[m * LDS_STRIDE + k0 * 32 + 16 + kh * 8];
                v16h a;
#pragma unroll
                for (int i = 0; i < 8; ++i) { a[i] = lo[i]; a[i + 8] = hi[i]; }
                c = __builtin_amdgcn_wmma_f32_16x16x32_f16(
                        false, a, false, Wrec[k0], (short)0, c, false, false);
            }
        }
        __syncthreads();   // all reads of old sS / sX complete before overwrite

        // ---- state = tanh(c); publish to LDS (f16) for next step ----
#pragma unroll
        for (int r = 0; r < 8; ++r) {
            const float y = fast_tanh(c[r]);
            st[r] = y;
            // D layout: VGPR r holds M = r + 8*kh, column = col
            sS[(r + 8 * kh) * LDS_STRIDE + col] = (_Float16)y;
        }
    }

    // ---- Write final state [B,H] in f32 ----
#pragma unroll
    for (int r = 0; r < 8; ++r) {
        const int row = wgbase + r + 8 * kh;
        out[(size_t)row * RNN_H + col] = st[r];
    }
}

extern "C" void kernel_launch(void* const* d_in, const int* in_sizes, int n_in,
                              void* d_out, int out_size, void* d_ws, size_t ws_size,
                              hipStream_t stream) {
    const float* x  = (const float*)d_in[0];   // [B,T,D]
    const float* w  = (const float*)d_in[1];   // [D,H]
    const float* sw = (const float*)d_in[2];   // [H,H]
    const float* b  = (const float*)d_in[3];   // [H]
    float* out = (float*)d_out;                // [B,H]

    dim3 grid(RNN_B / 16);   // 16 workgroups, one per 16-row batch block
    dim3 block(256);         // 8 wave32 waves -> 8 column tiles of H=128
    DenseRNN_elman_wmma_kernel<<<grid, block, 0, stream>>>(x, w, sw, b, out);
}
